// EGCFv2Model_9509057593697
// MI455X (gfx1250) — matrix operand order repaired and checked by reading, not compiled, not run on hardware
//
#include <hip/hip_runtime.h>

typedef __attribute__((ext_vector_type(16))) _Float16 v16h;
typedef __attribute__((ext_vector_type(8)))  float    v8f;

#define K_DIM 64   // node embedding dim
#define F_DIM 64   // edge feature dim

__device__ __forceinline__ float fast_sigmoid(float x) {
  // 1/(1+e^-x) with single v_rcp_f32 (avoid IEEE div expansion)
  return __builtin_amdgcn_rcpf(1.0f + __expf(-x));
}

// ---------------------------------------------------------------------------
// deg[n] += 1 for each incoming edge (segment_sum of ones over dst)
// ---------------------------------------------------------------------------
__global__ void deg_kernel(const int* __restrict__ dst, float* __restrict__ deg, int E) {
  int i = blockIdx.x * blockDim.x + threadIdx.x;
  for (; i < E; i += gridDim.x * blockDim.x)
    atomicAdd(&deg[dst[i]], 1.0f);
}

// ---------------------------------------------------------------------------
// norm[e] = rsqrt(max(deg[src],1) * max(deg[dst],1))
// ---------------------------------------------------------------------------
__global__ void norm_kernel(const int* __restrict__ src, const int* __restrict__ dst,
                            const float* __restrict__ deg, float* __restrict__ nrm, int E) {
  int i = blockIdx.x * blockDim.x + threadIdx.x;
  for (; i < E; i += gridDim.x * blockDim.x) {
    float ds = fmaxf(deg[src[i]], 1.0f);
    float dd = fmaxf(deg[dst[i]], 1.0f);
    nrm[i] = rsqrtf(ds * dd);
  }
}

// ---------------------------------------------------------------------------
// x = concat(Gu, Gi); acc = alpha0 * x  (alpha0 = 1)
// ---------------------------------------------------------------------------
__global__ void init_x_acc_kernel(const float* __restrict__ Gu, const float* __restrict__ Gi,
                                  float* __restrict__ x, float* __restrict__ acc,
                                  int nU64, int total) {
  int i = blockIdx.x * blockDim.x + threadIdx.x;
  for (; i < total; i += gridDim.x * blockDim.x) {
    float v = (i < nU64) ? Gu[i] : Gi[i - nU64];
    x[i]   = v;
    acc[i] = v;
  }
}

// acc += alpha * xn
__global__ void axpy_kernel(float* __restrict__ acc, const float* __restrict__ xn,
                            float alpha, int n) {
  int i = blockIdx.x * blockDim.x + threadIdx.x;
  for (; i < n; i += gridDim.x * blockDim.x)
    acc[i] = fmaf(alpha, xn[i], acc[i]);
}

// ---------------------------------------------------------------------------
// Fused edge-gated propagation layer.
// One wave32 processes 16-edge tiles:
//   gate(16x64) = sigmoid( Ge_tile(16x64) @ W_l(64x64) )   via 8x v_wmma_f32_16x16x32_f16
//   msg = gate * x[src] * norm  -> atomicAdd into xout[dst]
// The gate tile never touches memory (saves ~1 GB/layer of HBM round-trip).
//
// - B fragments live in LDS, fetched per nb-iteration via ds_load_b128; the
//   index is laundered through an empty asm so the loads can't be hoisted
//   (which previously caused 64-VGPR liveness + scratch spills).
// - Gather/scatter addresses are 32-bit element offsets (node*64+col fits in
//   i32; byte offset < 2^31) so vmem can use SGPR-base + 32-bit VGPR-offset
//   addressing with nb*64 folded into the immediate — no per-element 64-bit
//   address arithmetic.
// - All 8 gathers of an nb-iteration are issued before the WMMAs' results are
//   consumed, so they overlap the matrix op and LDS fragment latency.
// ---------------------------------------------------------------------------
__global__ void layer_kernel(const float* __restrict__ Ge,  // [E, F]
                             const float* __restrict__ Wl,  // [F, K]
                             const float* __restrict__ xin, // [N, K]
                             float* __restrict__ xout,      // [N, K] (pre-zeroed)
                             const float* __restrict__ nrm, // [E]
                             const int* __restrict__ src,
                             const int* __restrict__ dst,
                             int numTiles) {
  const int lane          = threadIdx.x & 31;
  const int waveInBlock   = threadIdx.x >> 5;
  const int wavesPerBlock = blockDim.x >> 5;
  const int gwave  = blockIdx.x * wavesPerBlock + waveInBlock;
  const int nWaves = gridDim.x * wavesPerBlock;

  const int col = lane & 15;   // B/C column, and A row (edge within tile) for loads
  const int hs  = lane >> 4;   // half-wave select

  // ---- Pack B fragments into LDS: Wfrag[kc][nb][lane] = this lane's v16h ----
  // B (32x16 f16) layout: lanes 0-15 hold K=0..15 of the chunk, lanes 16-31 hold
  // K=16..31; column N = lane&15; element e of the v16h is K = (lane>>4)*16 + e.
  __shared__ v16h Wfrag[2][4][32];     // 8 KB
  {
    _Float16* wp = (_Float16*)&Wfrag[0][0][0];
    const int total = 2 * 4 * 32 * 16;
    for (int idx = threadIdx.x; idx < total; idx += blockDim.x) {
      int e  = idx & 15;
      int ln = (idx >> 4) & 31;
      int nb = (idx >> 9) & 3;
      int kc = idx >> 11;
      int f  = kc * 32 + (ln >> 4) * 16 + e;   // GEMM K index = edge-feature f
      int cc = ln & 15;                        // output column within tile
      wp[idx] = (_Float16)Wl[f * K_DIM + nb * 16 + cc];
    }
  }
  __syncthreads();

  for (int t = gwave; t < numTiles; t += nWaves) {
    const int edgeBase = t * 16;

    // A fragments: 16x32 f16 per K-chunk. Lane row = lane&15.
    // Element e -> K = hs*16 + e  (+ kc*32)
    const float* gerow = Ge + (size_t)(edgeBase + col) * F_DIM;
    v16h afrag[2];
#pragma unroll
    for (int kc = 0; kc < 2; ++kc) {
#pragma unroll
      for (int p = 0; p < 8; ++p) {
        int k = kc * 32 + hs * 16 + p * 2;
        float2 f2 = *(const float2*)(gerow + k);
        afrag[kc][2 * p]     = (_Float16)f2.x;
        afrag[kc][2 * p + 1] = (_Float16)f2.y;
      }
    }

    // Edge metadata for the 8 rows this lane's C fragment covers (row = r+8*hs).
    // The 8 values are consecutive in memory -> two b128 loads per array.
    const int mbase = edgeBase + hs * 8;
    int4   sv0 = *(const int4*)(src + mbase),   sv1 = *(const int4*)(src + mbase + 4);
    int4   dv0 = *(const int4*)(dst + mbase),   dv1 = *(const int4*)(dst + mbase + 4);
    float4 nv0 = *(const float4*)(nrm + mbase), nv1 = *(const float4*)(nrm + mbase + 4);
    const int ss[8] = {sv0.x, sv0.y, sv0.z, sv0.w, sv1.x, sv1.y, sv1.z, sv1.w};
    const int dd[8] = {dv0.x, dv0.y, dv0.z, dv0.w, dv1.x, dv1.y, dv1.z, dv1.w};
    const float n8[8] = {nv0.x, nv0.y, nv0.z, nv0.w, nv1.x, nv1.y, nv1.z, nv1.w};
    // 32-bit element offsets (max N*K = 9.6M elements; byte offset < 2^31)
    int so8[8], do8[8];
#pragma unroll
    for (int r = 0; r < 8; ++r) {
      so8[r] = ss[r] * K_DIM + col;
      do8[r] = dd[r] * K_DIM + col;
    }

#pragma unroll
    for (int nb = 0; nb < 4; ++nb) {
      // Issue the 8 gathers up front so they overlap the LDS + WMMA latency.
      float xv[8];
#pragma unroll
      for (int r = 0; r < 8; ++r)
        xv[r] = xin[so8[r] + nb * 16];

      // Launder the fragment index so the LDS fragment loads stay in-loop.
      int fi = nb;
      asm volatile("" : "+v"(fi));
      v16h b0 = Wfrag[0][fi][lane];
      v16h b1 = Wfrag[1][fi][lane];

      v8f c = {};
      c = __builtin_amdgcn_wmma_f32_16x16x32_f16(false, afrag[0], false, b0,
                                                 (short)0, c, false, false);
      c = __builtin_amdgcn_wmma_f32_16x16x32_f16(false, afrag[1], false, b1,
                                                 (short)0, c, false, false);
#pragma unroll
      for (int r = 0; r < 8; ++r) {
        float gate = fast_sigmoid(c[r]);
        atomicAdd(&xout[do8[r] + nb * 16], gate * xv[r] * n8[r]);   // scatter-sum
      }
    }
  }
}

// Scalar fallback for a ragged tail (E % 16 != 0); not launched for the reference sizes.
__global__ void layer_tail_kernel(const float* __restrict__ Ge, const float* __restrict__ Wl,
                                  const float* __restrict__ xin, float* __restrict__ xout,
                                  const float* __restrict__ nrm, const int* __restrict__ src,
                                  const int* __restrict__ dst, int eStart, int E) {
  int idx = blockIdx.x * blockDim.x + threadIdx.x;
  int total = (E - eStart) * K_DIM;
  for (; idx < total; idx += gridDim.x * blockDim.x) {
    int e = eStart + idx / K_DIM;
    int k = idx % K_DIM;
    float a = 0.0f;
    for (int f = 0; f < F_DIM; ++f) a = fmaf(Ge[(size_t)e * F_DIM + f], Wl[f * K_DIM + k], a);
    float gate = fast_sigmoid(a);
    float msg  = gate * xin[(size_t)src[e] * K_DIM + k] * nrm[e];
    atomicAdd(&xout[(size_t)dst[e] * K_DIM + k], msg);
  }
}

// ---------------------------------------------------------------------------
// out[b] = dot(acc[user_idx[b]], acc[U + item_idx[b]])  -- 16 lanes per row,
// float4 loads + shfl_xor tree reduction.
// ---------------------------------------------------------------------------
__global__ void score_kernel(const float* __restrict__ acc, const int* __restrict__ uidx,
                             const int* __restrict__ iidx, float* __restrict__ out,
                             int B, int U) {
  int g = (int)((blockIdx.x * blockDim.x + threadIdx.x) >> 4);
  int l = threadIdx.x & 15;
  if (g >= B) return;
  const float* gu = acc + (size_t)uidx[g] * K_DIM;
  const float* gi = acc + (size_t)(U + iidx[g]) * K_DIM;
  float4 a = *(const float4*)(gu + l * 4);
  float4 b = *(const float4*)(gi + l * 4);
  float s = a.x * b.x + a.y * b.y + a.z * b.z + a.w * b.w;
#pragma unroll
  for (int off = 8; off; off >>= 1)
    s += __shfl_xor(s, off, 16);
  if (l == 0) out[g] = s;
}

// ---------------------------------------------------------------------------
extern "C" void kernel_launch(void* const* d_in, const int* in_sizes, int n_in,
                              void* d_out, int out_size, void* d_ws, size_t ws_size,
                              hipStream_t stream) {
  (void)n_in; (void)out_size; (void)ws_size;
  const float* Gu = (const float*)d_in[0];
  const float* Gi = (const float*)d_in[1];
  const float* Ge = (const float*)d_in[2];
  const float* W  = (const float*)d_in[3];
  const int* edge_index = (const int*)d_in[4];
  const int* user_idx   = (const int*)d_in[5];
  const int* item_idx   = (const int*)d_in[6];
  float* out = (float*)d_out;

  const int U = in_sizes[0] / K_DIM;
  const int I = in_sizes[1] / K_DIM;
  const int N = U + I;
  const int E = in_sizes[2] / F_DIM;
  const int L = in_sizes[3] / (F_DIM * K_DIM);
  const int B = in_sizes[5];
  const int* src = edge_index;
  const int* dst = edge_index + E;

  // carve scratch from d_ws
  char* p = (char*)d_ws;
  auto carve = [&](size_t bytes) -> void* {
    void* r = (void*)p;
    p += (bytes + 255) & ~(size_t)255;
    return r;
  };
  float* deg = (float*)carve((size_t)N * sizeof(float));
  float* nrm = (float*)carve((size_t)E * sizeof(float));
  float* xA  = (float*)carve((size_t)N * K_DIM * sizeof(float));
  float* xB  = (float*)carve((size_t)N * K_DIM * sizeof(float));
  float* acc = (float*)carve((size_t)N * K_DIM * sizeof(float));

  hipMemsetAsync(deg, 0, (size_t)N * sizeof(float), stream);
  deg_kernel<<<2048, 256, 0, stream>>>(dst, deg, E);
  norm_kernel<<<2048, 256, 0, stream>>>(src, dst, deg, nrm, E);
  init_x_acc_kernel<<<2048, 256, 0, stream>>>(Gu, Gi, xA, acc, U * K_DIM, N * K_DIM);

  float* xin = xA;
  float* xout = xB;
  const int numTiles = E / 16;
  for (int l = 0; l < L; ++l) {
    hipMemsetAsync(xout, 0, (size_t)N * K_DIM * sizeof(float), stream);
    layer_kernel<<<1024, 256, 0, stream>>>(Ge, W + (size_t)l * F_DIM * K_DIM,
                                           xin, xout, nrm, src, dst, numTiles);
    if (E % 16)
      layer_tail_kernel<<<64, 256, 0, stream>>>(Ge, W + (size_t)l * F_DIM * K_DIM,
                                                xin, xout, nrm, src, dst, numTiles * 16, E);
    axpy_kernel<<<2048, 256, 0, stream>>>(acc, xout, 1.0f / (float)(l + 2), N * K_DIM);
    float* t = xin; xin = xout; xout = t;
  }

  score_kernel<<<(B * 16 + 255) / 256, 256, 0, stream>>>(acc, user_idx, item_idx, out, B, U);
}
